// MLP_41334765256937
// MI455X (gfx1250) — matrix operand order repaired and checked by reading, not compiled, not run on hardware
//
#include <hip/hip_runtime.h>
#include <hip/hip_bf16.h>
#include <math.h>

typedef _Float16 v16h __attribute__((ext_vector_type(16)));
typedef _Float16 v8h  __attribute__((ext_vector_type(8)));
typedef _Float16 h2   __attribute__((ext_vector_type(2)));
typedef float    v8f  __attribute__((ext_vector_type(8)));

#define BATCH 8192
#define SEQ   50
#define EMB   512

// ---------------------------------------------------------------------------
// 1) Embedding bag: x[row, :] = mean_s E[ids[row,s], :]  -> f16
//    One block per batch row; 256 threads x float2 covers EMB=512.
//    ids[row*50+s] is block-uniform -> scalar loads; E reads fully coalesced.
// ---------------------------------------------------------------------------
__global__ __launch_bounds__(256)
void embed_mean_f16(const int* __restrict__ ids, const float* __restrict__ E,
                    _Float16* __restrict__ X) {
    const int row  = blockIdx.x;
    const int col0 = threadIdx.x * 2;
    const int* idr = ids + row * SEQ;
    float a0 = 0.f, a1 = 0.f;
    for (int s = 0; s < SEQ; ++s) {
        const float2 e = *(const float2*)(E + (size_t)idr[s] * EMB + col0);
        a0 += e.x; a1 += e.y;
    }
    const float inv = 1.0f / (float)SEQ;
    h2 o;
    o.x = (_Float16)(a0 * inv);
    o.y = (_Float16)(a1 * inv);
    *(h2*)(X + (size_t)row * EMB + col0) = o;
}

// ---------------------------------------------------------------------------
// 2) Weight prep: Wt[n, k] = (f16) W[k, n]   (K x N fp32 -> N x K f16)
// ---------------------------------------------------------------------------
__global__ __launch_bounds__(256)
void transpose_w_f16(const float* __restrict__ W, _Float16* __restrict__ Wt,
                     int K, int N) {
    int idx = blockIdx.x * blockDim.x + threadIdx.x;
    if (idx >= K * N) return;
    int n = idx / K;
    int k = idx - n * K;
    Wt[idx] = (_Float16)W[(size_t)k * N + n];
}

// ---------------------------------------------------------------------------
// 3) WMMA GEMM + bias + ReLU:  Y = relu(X @ W + b), X:[M,K] f16, Wt:[N,K] f16
//    Block = 128 threads = 4 waves stacked on M; each wave owns a 32x32
//    output tile = 2x2 grid of 16x16 WMMA tiles (4 independent v8f accums).
//    Per K-step of 32: 4 A-loads + 4 B-loads (b128) feed 4 independent
//    v_wmma_f32_16x16x32_f16 -> 2 loads/WMMA, no RAW chain between WMMAs.
//    A-frag (16x32 f16): lane l%16 = row, group g = lane/16; elems 0-7 at
//      K=k0+16g, elems 8-15 at K=k0+16g+8 (two contiguous 16B loads).
//    B-frag: Wt row N=l, 16 consecutive K at k0+16g (one 32B load).
//    C/D v8f: element r -> (M = r+8g, N = l).
// ---------------------------------------------------------------------------
__global__ __launch_bounds__(128)
void wmma_gemm_bias_relu(const _Float16* __restrict__ X,
                         const _Float16* __restrict__ Wt,
                         const float* __restrict__ bias,
                         _Float16* __restrict__ Y,
                         int M, int N, int K) {
    const int wave = threadIdx.x >> 5;
    const int lane = threadIdx.x & 31;
    const int g    = lane >> 4;   // half-wave group
    const int l    = lane & 15;
    const int tm   = (blockIdx.y * 4 + wave) * 2;  // M tile (16-row units): tm, tm+1
    const int tn   = blockIdx.x * 2;               // N tile (16-col units): tn, tn+1

    const _Float16* a0row = X  + (size_t)(tm * 16 + l) * K;
    const _Float16* a1row = X  + (size_t)(tm * 16 + 16 + l) * K;
    const _Float16* b0row = Wt + (size_t)(tn * 16 + l) * K;
    const _Float16* b1row = Wt + (size_t)(tn * 16 + 16 + l) * K;

    v8f c00 = {}, c01 = {}, c10 = {}, c11 = {};
    for (int k0 = 0; k0 < K; k0 += 32) {
        __builtin_prefetch(a0row + k0 + 128, 0, 0);
        __builtin_prefetch(a1row + k0 + 128, 0, 0);
        v8h  a0lo = *(const v8h*)(a0row + k0 + 16 * g);
        v8h  a0hi = *(const v8h*)(a0row + k0 + 16 * g + 8);
        v8h  a1lo = *(const v8h*)(a1row + k0 + 16 * g);
        v8h  a1hi = *(const v8h*)(a1row + k0 + 16 * g + 8);
        v16h b0   = *(const v16h*)(b0row + k0 + 16 * g);
        v16h b1   = *(const v16h*)(b1row + k0 + 16 * g);
        v16h a0, a1;
#pragma unroll
        for (int i = 0; i < 8; ++i) {
            a0[i] = a0lo[i]; a0[i + 8] = a0hi[i];
            a1[i] = a1lo[i]; a1[i + 8] = a1hi[i];
        }
        c00 = __builtin_amdgcn_wmma_f32_16x16x32_f16(false, a0, false, b0, (short)0, c00, false, false);
        c01 = __builtin_amdgcn_wmma_f32_16x16x32_f16(false, a0, false, b1, (short)0, c01, false, false);
        c10 = __builtin_amdgcn_wmma_f32_16x16x32_f16(false, a1, false, b0, (short)0, c10, false, false);
        c11 = __builtin_amdgcn_wmma_f32_16x16x32_f16(false, a1, false, b1, (short)0, c11, false, false);
    }

    const int   n0  = tn * 16 + l;
    const int   n1  = n0 + 16;
    const float bn0 = bias[n0];
    const float bn1 = bias[n1];
#pragma unroll
    for (int r = 0; r < 8; ++r) {
        const int m0 = tm * 16 + r + 8 * g;
        const int m1 = m0 + 16;
        float v;
        v = c00[r] + bn0; v = v > 0.f ? v : 0.f; Y[(size_t)m0 * N + n0] = (_Float16)v;
        v = c01[r] + bn1; v = v > 0.f ? v : 0.f; Y[(size_t)m0 * N + n1] = (_Float16)v;
        v = c10[r] + bn0; v = v > 0.f ? v : 0.f; Y[(size_t)m1 * N + n0] = (_Float16)v;
        v = c11[r] + bn1; v = v > 0.f ? v : 0.f; Y[(size_t)m1 * N + n1] = (_Float16)v;
    }
}

// ---------------------------------------------------------------------------
// 4) Final layer 64 -> 3 + softmax, fp32 out. One thread per row.
// ---------------------------------------------------------------------------
__global__ __launch_bounds__(256)
void final_softmax(const _Float16* __restrict__ X, const float* __restrict__ W,
                   const float* __restrict__ b, float* __restrict__ out, int M) {
    int row = blockIdx.x * blockDim.x + threadIdx.x;
    if (row >= M) return;
    const _Float16* xr = X + (size_t)row * 64;
    float a0 = b[0], a1 = b[1], a2 = b[2];
#pragma unroll 8
    for (int k = 0; k < 64; ++k) {
        const float x = (float)xr[k];
        a0 = fmaf(x, W[k * 3 + 0], a0);
        a1 = fmaf(x, W[k * 3 + 1], a1);
        a2 = fmaf(x, W[k * 3 + 2], a2);
    }
    const float mx = fmaxf(a0, fmaxf(a1, a2));
    const float e0 = __expf(a0 - mx), e1 = __expf(a1 - mx), e2 = __expf(a2 - mx);
    const float inv = 1.0f / (e0 + e1 + e2);
    out[row * 3 + 0] = e0 * inv;
    out[row * 3 + 1] = e1 * inv;
    out[row * 3 + 2] = e2 * inv;
}

// ---------------------------------------------------------------------------
// Launch
// ---------------------------------------------------------------------------
extern "C" void kernel_launch(void* const* d_in, const int* in_sizes, int n_in,
                              void* d_out, int out_size, void* d_ws, size_t ws_size,
                              hipStream_t stream) {
    (void)in_sizes; (void)n_in; (void)out_size; (void)ws_size;

    const int*   ids = (const int*)  d_in[0];
    const float* E   = (const float*)d_in[1];
    const float* W0  = (const float*)d_in[2];  const float* b0 = (const float*)d_in[3];
    const float* W1  = (const float*)d_in[4];  const float* b1 = (const float*)d_in[5];
    const float* W2  = (const float*)d_in[6];  const float* b2 = (const float*)d_in[7];
    const float* W3  = (const float*)d_in[8];  const float* b3 = (const float*)d_in[9];
    const float* W4  = (const float*)d_in[10]; const float* b4 = (const float*)d_in[11];
    float* out = (float*)d_out;

    // Workspace layout: two ping-pong activation buffers (f16, widest layer)
    // + f16 transposed weights.  Total ~17.6 MB.
    char* ws = (char*)d_ws;
    size_t off = 0;
    _Float16* xA  = (_Float16*)(ws + off); off += (size_t)BATCH * 512 * sizeof(_Float16);
    _Float16* xB  = (_Float16*)(ws + off); off += (size_t)BATCH * 512 * sizeof(_Float16);
    _Float16* wt0 = (_Float16*)(ws + off); off += (size_t)512 * 512 * sizeof(_Float16);
    _Float16* wt1 = (_Float16*)(ws + off); off += (size_t)256 * 512 * sizeof(_Float16);
    _Float16* wt2 = (_Float16*)(ws + off); off += (size_t)128 * 256 * sizeof(_Float16);
    _Float16* wt3 = (_Float16*)(ws + off); off += (size_t)64  * 128 * sizeof(_Float16);

    // 1) embedding bag (HBM-bound, ~840 MB gather -> the real time floor)
    embed_mean_f16<<<BATCH, 256, 0, stream>>>(ids, E, xA);

    // 2) weight transpose+convert (tiny)
    auto tgrid = [](int n) { return dim3((n + 255) / 256); };
    transpose_w_f16<<<tgrid(512 * 512), 256, 0, stream>>>(W0, wt0, 512, 512);
    transpose_w_f16<<<tgrid(512 * 256), 256, 0, stream>>>(W1, wt1, 512, 256);
    transpose_w_f16<<<tgrid(256 * 128), 256, 0, stream>>>(W2, wt2, 256, 128);
    transpose_w_f16<<<tgrid(128 * 64),  256, 0, stream>>>(W3, wt3, 128, 64);

    // 3) WMMA MLP chain; each block computes 128(M) x 32(N) of output.
    wmma_gemm_bias_relu<<<dim3(512 / 32, BATCH / 128), 128, 0, stream>>>(xA, wt0, b0, xB, BATCH, 512, 512);
    wmma_gemm_bias_relu<<<dim3(256 / 32, BATCH / 128), 128, 0, stream>>>(xB, wt1, b1, xA, BATCH, 256, 512);
    wmma_gemm_bias_relu<<<dim3(128 / 32, BATCH / 128), 128, 0, stream>>>(xA, wt2, b2, xB, BATCH, 128, 256);
    wmma_gemm_bias_relu<<<dim3(64 / 32,  BATCH / 128), 128, 0, stream>>>(xB, wt3, b3, xA, BATCH, 64, 128);

    // 4) 64->3 + softmax, fp32 output
    final_softmax<<<dim3(BATCH / 256), 256, 0, stream>>>(xA, W4, b4, out, BATCH);
}
